// cheb_conv_withSAt_59493886984745
// MI455X (gfx1250) — matrix operand order repaired and checked by reading, not compiled, not run on hardware
//
#include <hip/hip_runtime.h>
#include <hip/hip_bf16.h>

// Problem constants
#define BB 8
#define NN 1024
#define FF 64
#define TT 32
#define KC 3
#define OO 64
#define KJ (KC * NN)      // 3072  reduction dim of big GEMM
#define C2 (OO * TT)      // 2048  N dim of big GEMM

typedef __attribute__((ext_vector_type(16))) __bf16 v16bf;
typedef __attribute__((ext_vector_type(8)))  float  v8f;

union AFrag { v16bf v; unsigned int u[8]; unsigned short s[16]; uint4 q[2]; };

#define AS1 __attribute__((address_space(1)))
#define AS3 __attribute__((address_space(3)))

#ifndef __has_builtin
#define __has_builtin(x) 0
#endif
#if __has_builtin(__builtin_amdgcn_global_load_async_to_lds_b128)
#define HAVE_ASYNC_LDS 1
#else
#define HAVE_ASYNC_LDS 0
#endif

// Pointee type the builtin expects: clang vector of 4 ints.
typedef __attribute__((__vector_size__(4 * sizeof(int)))) int v4i_t;

// 16-byte global -> LDS copy: async on CDNA5 (tracked by ASYNCcnt),
// plain b128 load + ds_store fallback otherwise.
__device__ __forceinline__ void cp16(unsigned short* dst_lds,
                                     const unsigned short* src_g) {
#if HAVE_ASYNC_LDS
    __builtin_amdgcn_global_load_async_to_lds_b128(
        (AS1 v4i_t*)(size_t)src_g,
        (AS3 v4i_t*)(unsigned)(size_t)dst_lds,
        0, 0);
#else
    *reinterpret_cast<uint4*>(dst_lds) =
        *reinterpret_cast<const uint4*>(src_g);
#endif
}

__device__ __forceinline__ void wait_async() {
#if HAVE_ASYNC_LDS
#if __has_builtin(__builtin_amdgcn_s_wait_asynccnt)
    __builtin_amdgcn_s_wait_asynccnt(0);
#else
    asm volatile("s_wait_asynccnt 0x0" ::: "memory");
#endif
#endif
}

__device__ __forceinline__ unsigned short f2bf(float f) {
    unsigned int u = __float_as_uint(f);
    unsigned int r = (u + 0x7FFFu + ((u >> 16) & 1u)) >> 16;  // RNE
    return (unsigned short)r;
}

// ---------------------------------------------------------------------------
// Kernel 1: ThetaT[(k*64+o)][f] = bf16(Theta[k][f][o])   (192 x 64)
// ---------------------------------------------------------------------------
__global__ __launch_bounds__(256) void theta_k(const float* __restrict__ Theta,
                                               unsigned short* __restrict__ thetaT) {
    for (int e = threadIdx.x; e < 192 * 64; e += 256) {
        int m = e >> 6, f = e & 63;
        int k = m >> 6, o = m & 63;
        thetaT[e] = f2bf(Theta[k * (FF * OO) + f * OO + o]);
    }
}

// ---------------------------------------------------------------------------
// Kernel 2: W[b][i][k*N+j] = bf16(cheb[k][j][i] * att[b][j][i])
// 64x64 LDS tile transpose: coalesced reads and writes.
// ---------------------------------------------------------------------------
__global__ __launch_bounds__(256) void wbuild_k(const float* __restrict__ cheb,
                                                const float* __restrict__ att,
                                                unsigned short* __restrict__ W) {
    __shared__ unsigned short tile[64][72];
    int bk = blockIdx.z;
    int b = bk / KC, k = bk % KC;
    int i0 = blockIdx.x * 64, j0 = blockIdx.y * 64;
    int tid = threadIdx.x;
    int rl = tid >> 6, cl = tid & 63;
    const float* cb = cheb + ((size_t)k * NN + j0) * NN + i0;
    const float* ab = att  + ((size_t)b * NN + j0) * NN + i0;
    for (int s = 0; s < 16; ++s) {
        int j = rl + s * 4;
        tile[j][cl] = f2bf(cb[(size_t)j * NN + cl] * ab[(size_t)j * NN + cl]);
    }
    __syncthreads();
    unsigned short* wb = W + ((size_t)b * NN + i0) * KJ + (size_t)k * NN + j0;
    for (int s = 0; s < 16; ++s) {
        int i = rl + s * 4;
        wb[(size_t)i * KJ + cl] = tile[cl][i];
    }
}

// ---------------------------------------------------------------------------
// Kernel 3: Y[b][k*N+j][o*T+t] = bf16( sum_f x[b,j,f,t] * Theta[k,f,o] )
// One block (128 thr = 4 waves) per (b,j). GEMM M=192, N=32, K=64 via WMMA.
// x staged transposed ([t][f]) so B fragments are 2x ds_load_b128.
// ---------------------------------------------------------------------------
__global__ __launch_bounds__(128) void ybuild_k(const float* __restrict__ x,
                                                const unsigned short* __restrict__ thetaT,
                                                unsigned short* __restrict__ Y) {
    __shared__ alignas(16) unsigned short lth[192 * 64]; // A: [(k,o)][f]
    __shared__ alignas(16) unsigned short lxT[32 * 64];  // B^T: [t][f]
    int bj = blockIdx.x;
    int b = bj >> 10, j = bj & (NN - 1);
    int tid = threadIdx.x, lane = tid & 31, w = tid >> 5;

    {   // stage ThetaT (24 KB) as b128 chunks
        const uint4* ts = reinterpret_cast<const uint4*>(thetaT);
        uint4* td = reinterpret_cast<uint4*>(lth);
        for (int s = 0; s < 12; ++s) td[tid + s * 128] = ts[tid + s * 128];
    }
    {   // stage x[b][j] (fp32 -> bf16), transpose [f][t] -> [t][f]
        const float* xg = x + ((size_t)b * NN + j) * (FF * TT);
        for (int s = 0; s < 4; ++s) {
            int e = tid * 4 + s * 512;       // e = f*32 + t, t multiple of 4
            int f = e >> 5, t = e & 31;
            float4 v = *reinterpret_cast<const float4*>(xg + e);
            lxT[(t + 0) * 64 + f] = f2bf(v.x);
            lxT[(t + 1) * 64 + f] = f2bf(v.y);
            lxT[(t + 2) * 64 + f] = f2bf(v.z);
            lxT[(t + 3) * 64 + f] = f2bf(v.w);
        }
    }
    __syncthreads();

    v8f acc[3][2];
    for (int a = 0; a < 3; ++a)
        for (int n = 0; n < 2; ++n)
            for (int r = 0; r < 8; ++r) acc[a][n][r] = 0.0f;

    int half = (lane & 16) ? 1 : 0;
    for (int ks = 0; ks < 2; ++ks) {
        AFrag af[3];
        for (int mi = 0; mi < 3; ++mi) {
            int m = (w * 3 + mi) * 16 + (lane & 15);
            const unsigned short* p = lth + m * 64 + half * 8 + ks * 32;
            af[mi].q[0] = *reinterpret_cast<const uint4*>(p);
            af[mi].q[1] = *reinterpret_cast<const uint4*>(p + 16);
        }
        AFrag bfr[2];
        for (int nt = 0; nt < 2; ++nt) {
            int t = nt * 16 + (lane & 15);
            const unsigned short* p = lxT + t * 64 + half * 8 + ks * 32;
            bfr[nt].q[0] = *reinterpret_cast<const uint4*>(p);
            bfr[nt].q[1] = *reinterpret_cast<const uint4*>(p + 16);
        }
        for (int mi = 0; mi < 3; ++mi)
            for (int nt = 0; nt < 2; ++nt)
                acc[mi][nt] = __builtin_amdgcn_wmma_f32_16x16x32_bf16(
                    false, af[mi].v, false, bfr[nt].v, (short)0, acc[mi][nt],
                    false, false);
    }

    for (int mi = 0; mi < 3; ++mi) {
        int mt = w * 3 + mi;
        int k = mt >> 2, ob = (mt & 3) * 16;
        size_t row = ((size_t)b * KJ + (size_t)k * NN + j) * C2;
        for (int nt = 0; nt < 2; ++nt) {
            int t = nt * 16 + (lane & 15);
            for (int r = 0; r < 8; ++r) {
                int o = ob + r + half * 8;
                Y[row + o * TT + t] = f2bf(acc[mi][nt][r]);
            }
        }
    }
}

// ---------------------------------------------------------------------------
// Kernel 3b: Yt[b][oc][kj] = Y[b][kj][oc]  (B operand transposed so the big
// GEMM's B fragments are K-contiguous in LDS)
// ---------------------------------------------------------------------------
__global__ __launch_bounds__(256) void ytrans_k(const unsigned short* __restrict__ Y,
                                                unsigned short* __restrict__ Yt) {
    __shared__ unsigned short tile[64][72];
    int b = blockIdx.z;
    int oc0 = blockIdx.x * 64, kj0 = blockIdx.y * 64;
    int tid = threadIdx.x;
    int rl = tid >> 6, cl = tid & 63;
    const unsigned short* ys = Y + ((size_t)b * KJ + kj0) * C2 + oc0;
    for (int s = 0; s < 16; ++s) {
        int r = rl + s * 4;
        tile[r][cl] = ys[(size_t)r * C2 + cl];
    }
    __syncthreads();
    unsigned short* yd = Yt + ((size_t)b * C2 + oc0) * KJ + kj0;
    for (int s = 0; s < 16; ++s) {
        int r = rl + s * 4;
        yd[(size_t)r * KJ + cl] = tile[cl][r];
    }
}

// ---------------------------------------------------------------------------
// Kernel 4: out[b][i][oc] = relu( sum_kj W[b][i][kj] * Yt[b][oc][kj] )
// Per b: 1024 x 2048 x 3072 GEMM. Block tile 128x128, K-step 32.
// Double-buffered LDS; staging via async global->LDS b128.
// 8 waves (2 x 4): each wave 64x32 slice = 4x2 WMMA accumulator tiles.
// ---------------------------------------------------------------------------
__global__ __launch_bounds__(256) void gemm_k(const unsigned short* __restrict__ Wb,
                                              const unsigned short* __restrict__ Yt,
                                              float* __restrict__ out) {
    __shared__ alignas(16) unsigned short As[2][128 * 32]; // [i][k]
    __shared__ alignas(16) unsigned short Bs[2][128 * 32]; // [n][k]
    int b  = blockIdx.z;
    int i0 = blockIdx.y * 128;
    int n0 = blockIdx.x * 128;
    int tid = threadIdx.x, lane = tid & 31, wave = tid >> 5;
    int wm = wave >> 2;                 // 0..1  (64-row slice)
    int wn = wave & 3;                  // 0..3  (32-col slice)
    int half = (lane & 16) ? 1 : 0;

    v8f acc[4][2];
    for (int a = 0; a < 4; ++a)
        for (int n = 0; n < 2; ++n)
            for (int r = 0; r < 8; ++r) acc[a][n][r] = 0.0f;

    const unsigned short* Arow = Wb + ((size_t)b * NN + i0) * KJ; // +row*KJ
    const unsigned short* Brow = Yt + ((size_t)b * C2 + n0) * KJ; // +row*KJ

    // Each thread stages 2 A-chunks + 2 B-chunks (b128 each) per K-step.
    auto stage = [&](int buf, int kk0) {
        for (int c = 0; c < 2; ++c) {
            int ch  = tid + c * 256;            // 0..511
            int row = ch >> 2, off = (ch & 3) * 8;
            cp16(&As[buf][row * 32 + off], Arow + (size_t)row * KJ + kk0 + off);
            cp16(&Bs[buf][row * 32 + off], Brow + (size_t)row * KJ + kk0 + off);
        }
    };

    stage(0, 0);
    wait_async();
    __syncthreads();

    int buf = 0;
    for (int s = 0; s < KJ / 32; ++s) {
        int kk_next = (s + 1) * 32;
        if (kk_next < KJ) stage(buf ^ 1, kk_next);   // prefetch next slab

        AFrag af[4];
        for (int mt = 0; mt < 4; ++mt) {
            int m = wm * 64 + mt * 16 + (lane & 15);
            const unsigned short* p = As[buf] + m * 32 + half * 8;
            af[mt].q[0] = *reinterpret_cast<const uint4*>(p);
            af[mt].q[1] = *reinterpret_cast<const uint4*>(p + 16);
        }
        AFrag bfr[2];
        for (int nt = 0; nt < 2; ++nt) {
            int n = wn * 32 + nt * 16 + (lane & 15);
            const unsigned short* p = Bs[buf] + n * 32 + half * 8;
            bfr[nt].q[0] = *reinterpret_cast<const uint4*>(p);
            bfr[nt].q[1] = *reinterpret_cast<const uint4*>(p + 16);
        }
        for (int mt = 0; mt < 4; ++mt)
            for (int nt = 0; nt < 2; ++nt)
                acc[mt][nt] = __builtin_amdgcn_wmma_f32_16x16x32_bf16(
                    false, af[mt].v, false, bfr[nt].v, (short)0, acc[mt][nt],
                    false, false);

        wait_async();       // next slab landed in LDS
        __syncthreads();    // everyone done reading buf / writing buf^1
        buf ^= 1;
    }

    // fused ReLU epilogue, fp32 store (out[b][i][o*T+t])
    for (int mt = 0; mt < 4; ++mt) {
        for (int nt = 0; nt < 2; ++nt) {
            int col = n0 + wn * 32 + nt * 16 + (lane & 15);
            for (int r = 0; r < 8; ++r) {
                int row = i0 + wm * 64 + mt * 16 + r + half * 8;
                out[((size_t)b * NN + row) * C2 + col] = fmaxf(acc[mt][nt][r], 0.0f);
            }
        }
    }
}

// ---------------------------------------------------------------------------
extern "C" void kernel_launch(void* const* d_in, const int* in_sizes, int n_in,
                              void* d_out, int out_size, void* d_ws, size_t ws_size,
                              hipStream_t stream) {
    const float* x     = (const float*)d_in[0];  // (B,N,F,T)
    const float* att   = (const float*)d_in[1];  // (B,N,N)
    const float* cheb  = (const float*)d_in[2];  // (K,N,N)
    const float* Theta = (const float*)d_in[3];  // (K,F,O)
    float* out = (float*)d_out;                  // (B,N,O,T) fp32

    char* ws = (char*)d_ws;
    const size_t wBytes = (size_t)BB * NN * KJ * 2;   // 50.3 MB
    const size_t yBytes = (size_t)BB * KJ * C2 * 2;   // 100.7 MB
    unsigned short* thetaT = (unsigned short*)ws;                     // 24.6 KB
    unsigned short* W  = (unsigned short*)(ws + 32768);
    unsigned short* Y  = (unsigned short*)(ws + 32768 + wBytes);
    unsigned short* Yt = (unsigned short*)(ws + 32768 + wBytes + yBytes);

    theta_k<<<dim3(1), 256, 0, stream>>>(Theta, thetaT);
    wbuild_k<<<dim3(16, 16, BB * KC), 256, 0, stream>>>(cheb, att, W);
    ybuild_k<<<dim3(BB * NN), 128, 0, stream>>>(x, thetaT, Y);
    ytrans_k<<<dim3(C2 / 64, KJ / 64, BB), 256, 0, stream>>>(Y, Yt);
    gemm_k<<<dim3(C2 / 128, NN / 128, BB), 256, 0, stream>>>(W, Yt, out);
}
